// Decoder_83167746719781
// MI455X (gfx1250) — matrix operand order repaired and checked by reading, not compiled, not run on hardware
//
#include <hip/hip_runtime.h>
#include <hip/hip_bf16.h>

// ---------------- types ----------------
typedef __bf16 bf16_t;
typedef __attribute__((ext_vector_type(16))) __bf16 v16bf;
typedef __attribute__((ext_vector_type(8)))  __bf16 v8bf;
typedef __attribute__((ext_vector_type(8)))  float  v8f;

#define BB   16      // batch
#define TT   800     // encoder length
#define EP   512     // eprojs
#define DU   1024    // dunits
#define AT   512     // att_dim
#define OD   5000    // odim
#define ODP  5008    // odim padded to multiple of 16
#define LM   96      // lmax
#define ST   97      // decoder steps (L+1)
#define NEG_INF (-__builtin_inff())

// ---------------- helpers ----------------
__device__ __forceinline__ v8f vzero8() {
  v8f z = {0.f,0.f,0.f,0.f,0.f,0.f,0.f,0.f};
  return z;
}

__device__ __forceinline__ float sigf(float x) { return 1.f / (1.f + __expf(-x)); }

__device__ __forceinline__ float wave_sum(float v) {
  for (int o = 16; o > 0; o >>= 1) v += __shfl_xor(v, o, 32);
  return v;
}
__device__ __forceinline__ float wave_max(float v) {
  for (int o = 16; o > 0; o >>= 1) v = fmaxf(v, __shfl_xor(v, o, 32));
  return v;
}

// A fragment: 16x32 bf16 tile from row-major A (pre-offset to tile row 0).
// lane 0-15: M=lane, K chunks [k0..k0+7],[k0+16..k0+23]
// lane 16-31: M=lane-16, K chunks [k0+8..k0+15],[k0+24..k0+31]
__device__ __forceinline__ v16bf load_a_tile(const bf16_t* __restrict__ A, int lda,
                                             int k0, int lane) {
  const int m  = lane & 15;
  const int kb = k0 + ((lane & 16) ? 8 : 0);
  const bf16_t* p = A + (long)m * lda + kb;
  v8bf lo = *(const v8bf*)(p);
  v8bf hi = *(const v8bf*)(p + 16);
  return __builtin_shufflevector(lo, hi, 0,1,2,3,4,5,6,7,8,9,10,11,12,13,14,15);
}

// B fragment: B[k][n] = W[n0+n][k] with W row-major (pre-offset to row n0).
// lane 0-15: N=lane, K=[k0..k0+7],[k0+16..k0+23]; lane 16-31: N=lane-16, K=[k0+8..k0+15],[k0+24..k0+31]
__device__ __forceinline__ v16bf load_b_tile(const bf16_t* __restrict__ W, int ldw,
                                             int k0, int lane) {
  const int n  = lane & 15;
  const int kb = k0 + ((lane & 16) ? 8 : 0);
  const bf16_t* p = W + (long)n * ldw + kb;
  v8bf lo = *(const v8bf*)(p);
  v8bf hi = *(const v8bf*)(p + 16);
  return __builtin_shufflevector(lo, hi, 0,1,2,3,4,5,6,7,8,9,10,11,12,13,14,15);
}

__device__ __forceinline__ v8f wmma_bf16(v16bf a, v16bf b, v8f c) {
  return __builtin_amdgcn_wmma_f32_16x16x32_bf16(false, a, false, b, (short)0, c,
                                                 false, false);
}

// online log-sum-exp merge
__device__ __forceinline__ void lse_merge(float& m1, float& s1, float m2, float s2) {
  if (!(m2 > NEG_INF)) return;
  if (!(m1 > NEG_INF)) { m1 = m2; s1 = s2; return; }
  if (m2 > m1) { s1 = s1 * __expf(m1 - m2) + s2; m1 = m2; }
  else         { s1 += s2 * __expf(m2 - m1); }
}

// ---------------- small utility kernels ----------------
__global__ void k_cvt(const float* __restrict__ in, bf16_t* __restrict__ out, int n) {
  int i = blockIdx.x * 256 + threadIdx.x;
  if (i < n) out[i] = (bf16_t)in[i];
}

__global__ void k_wout_pad(const float* __restrict__ W, bf16_t* __restrict__ out) {
  int i = blockIdx.x * 256 + threadIdx.x;
  if (i >= ODP * DU) return;
  int r = i / DU;
  out[i] = (r < OD) ? (bf16_t)W[i] : (bf16_t)0.f;
}

__global__ void k_zero32(unsigned int* __restrict__ p, int nwords) {
  int i = blockIdx.x * 256 + threadIdx.x;
  if (i < nwords) p[i] = 0u;
}

// ey_all[s][b][:] = embed[token(s,b)]  (bf16)
__global__ void k_embed_gather(const float* __restrict__ embed,
                               const int* __restrict__ ys_pad,
                               bf16_t* __restrict__ ey_all) {
  const int row = blockIdx.x;          // row = s*16 + b
  const int s = row / BB, b = row % BB;
  const int tok = (s == 0) ? (OD - 1) : ys_pad[b * LM + (s - 1)];
  const float* src = embed + (long)tok * DU;
  bf16_t* dst = ey_all + (long)row * DU;
  for (int d = threadIdx.x; d < DU; d += 256) dst[d] = (bf16_t)src[d];
}

// ---------------- pre_enc = hs @ W_enc^T + b_enc (once) ----------------
__global__ void k_pre_enc(const bf16_t* __restrict__ hs_bf,
                          const bf16_t* __restrict__ Wenc,
                          const float* __restrict__ benc,
                          float* __restrict__ pre) {
  const int nt = blockIdx.x;           // 0..31
  const int mt = blockIdx.y;           // 0..799
  const int lane = threadIdx.x;
  const bf16_t* A  = hs_bf + (long)mt * 16 * EP;
  const bf16_t* Wp = Wenc + (long)nt * 16 * EP;
  v8f acc = vzero8();
  for (int k = 0; k < EP; k += 32)
    acc = wmma_bf16(load_a_tile(A, EP, k, lane), load_b_tile(Wp, EP, k, lane), acc);
  const int n = lane & 15;
  const int mh = (lane & 16) ? 8 : 0;
  const float bias = benc[nt * 16 + n];
#pragma unroll
  for (int r = 0; r < 8; ++r)
    pre[((long)mt * 16 + mh + r) * AT + nt * 16 + n] = acc[r] + bias;
}

// ---------------- per-step: dec_proj = z0 @ W_dec^T ----------------
__global__ void k_dec_proj(const bf16_t* __restrict__ z0,
                           const bf16_t* __restrict__ Wdec,
                           float* __restrict__ dp) {
  const int nt = blockIdx.x;           // 0..31
  const int lane = threadIdx.x;
  const bf16_t* Wp = Wdec + (long)nt * 16 * DU;
  v8f acc = vzero8();
  for (int k = 0; k < DU; k += 32)
    acc = wmma_bf16(load_a_tile(z0, DU, k, lane), load_b_tile(Wp, DU, k, lane), acc);
  const int n = lane & 15;
  const int mh = (lane & 16) ? 8 : 0;
#pragma unroll
  for (int r = 0; r < 8; ++r)
    dp[(mh + r) * AT + nt * 16 + n] = acc[r];
}

// ---------------- per-step: additive attention ----------------
__global__ void k_attention(const float* __restrict__ pre, const float* __restrict__ dp,
                            const float* __restrict__ gvec,
                            const float* __restrict__ hs_pad,
                            const int* __restrict__ hlens,
                            bf16_t* __restrict__ attc) {
  const int b = blockIdx.x, tid = threadIdx.x;
  const int wid = tid >> 5, lane = tid & 31;
  __shared__ float s_dp[AT], s_g[AT], s_e[TT], s_red[8];
  const int hlen = hlens[b];
  for (int a = tid; a < AT; a += 256) { s_dp[a] = dp[b * AT + a]; s_g[a] = gvec[a]; }
  __syncthreads();
  const float* preb = pre + (long)b * TT * AT;
  float lmax = NEG_INF;
  for (int t = wid; t < TT; t += 8) {
    const float* pr = preb + (long)t * AT;
    float p = 0.f;
    for (int a = lane; a < AT; a += 32) p += s_g[a] * tanhf(pr[a] + s_dp[a]);
    p = wave_sum(p);
    float e = 2.0f * p;                       // SCALING = 2.0
    if (t >= hlen) e = NEG_INF;
    if (lane == 0) s_e[t] = e;
    lmax = fmaxf(lmax, e);
  }
  lmax = wave_max(lmax);
  if (lane == 0) s_red[wid] = lmax;
  __syncthreads();
  float bmax = NEG_INF;
  for (int i = 0; i < 8; ++i) bmax = fmaxf(bmax, s_red[i]);
  __syncthreads();                            // done reading s_red
  float lsum = 0.f;
  for (int t = tid; t < TT; t += 256) {
    float e = s_e[t];
    float w = (e > NEG_INF) ? __expf(e - bmax) : 0.f;
    s_e[t] = w;
    lsum += w;
  }
  lsum = wave_sum(lsum);
  if (lane == 0) s_red[wid] = lsum;
  __syncthreads();                            // s_e + s_red visible
  float bsum = 0.f;
  for (int i = 0; i < 8; ++i) bsum += s_red[i];
  const float inv = 1.f / bsum;
  for (int d = tid; d < EP; d += 256) {
    const float* h = hs_pad + (long)b * TT * EP + d;
    float sacc = 0.f;
    for (int t = 0; t < TT; ++t) sacc += s_e[t] * h[(long)t * EP];
    attc[b * EP + d] = (bf16_t)(sacc * inv);
  }
}

// ---------------- per-step: LSTM layer 0 ----------------
// gates = [ey ; attc] @ W_ih0^T + z0 @ W_hh0^T + b ; pointwise update.
__global__ void k_lstm0(const bf16_t* __restrict__ ey, const bf16_t* __restrict__ attc,
                        const bf16_t* __restrict__ z0c,
                        const bf16_t* __restrict__ Wih, const bf16_t* __restrict__ Whh,
                        const float* __restrict__ bih, const float* __restrict__ bhh,
                        float* __restrict__ c0, bf16_t* __restrict__ z0n) {
  const int u0 = blockIdx.x * 16;
  const int lane = threadIdx.x;
  const bf16_t* Wi[4]; const bf16_t* Wh[4];
#pragma unroll
  for (int g = 0; g < 4; ++g) {
    Wi[g] = Wih + (long)(g * DU + u0) * (DU + EP);
    Wh[g] = Whh + (long)(g * DU + u0) * DU;
  }
  v8f acc[4];
#pragma unroll
  for (int g = 0; g < 4; ++g) acc[g] = vzero8();
  for (int k = 0; k < DU; k += 32) {                 // ey part (K=1024)
    v16bf a = load_a_tile(ey, DU, k, lane);
#pragma unroll
    for (int g = 0; g < 4; ++g)
      acc[g] = wmma_bf16(a, load_b_tile(Wi[g], DU + EP, k, lane), acc[g]);
  }
  for (int k = 0; k < EP; k += 32) {                 // attc part (K=512)
    v16bf a = load_a_tile(attc, EP, k, lane);
#pragma unroll
    for (int g = 0; g < 4; ++g)
      acc[g] = wmma_bf16(a, load_b_tile(Wi[g] + DU, DU + EP, k, lane), acc[g]);
  }
  for (int k = 0; k < DU; k += 32) {                 // recurrent part (K=1024)
    v16bf a = load_a_tile(z0c, DU, k, lane);
#pragma unroll
    for (int g = 0; g < 4; ++g)
      acc[g] = wmma_bf16(a, load_b_tile(Wh[g], DU, k, lane), acc[g]);
  }
  const int n = u0 + (lane & 15);
  const int mh = (lane & 16) ? 8 : 0;
#pragma unroll
  for (int r = 0; r < 8; ++r) {
    const int m = mh + r;
    float gi = acc[0][r] + bih[n]          + bhh[n];
    float gf = acc[1][r] + bih[DU + n]     + bhh[DU + n];
    float gg = acc[2][r] + bih[2 * DU + n] + bhh[2 * DU + n];
    float go = acc[3][r] + bih[3 * DU + n] + bhh[3 * DU + n];
    const long idx = (long)m * DU + n;
    float cn = sigf(gf) * c0[idx] + sigf(gi) * tanhf(gg);
    c0[idx] = cn;
    z0n[idx] = (bf16_t)(sigf(go) * tanhf(cn));
  }
}

// ---------------- per-step: LSTM layer 1 ----------------
__global__ void k_lstm1(const bf16_t* __restrict__ z0n, const bf16_t* __restrict__ z1c,
                        const bf16_t* __restrict__ Wih, const bf16_t* __restrict__ Whh,
                        const float* __restrict__ bih, const float* __restrict__ bhh,
                        float* __restrict__ c1, bf16_t* __restrict__ z1n,
                        bf16_t* __restrict__ zall_step) {
  const int u0 = blockIdx.x * 16;
  const int lane = threadIdx.x;
  const bf16_t* Wi[4]; const bf16_t* Wh[4];
#pragma unroll
  for (int g = 0; g < 4; ++g) {
    Wi[g] = Wih + (long)(g * DU + u0) * DU;
    Wh[g] = Whh + (long)(g * DU + u0) * DU;
  }
  v8f acc[4];
#pragma unroll
  for (int g = 0; g < 4; ++g) acc[g] = vzero8();
  for (int k = 0; k < DU; k += 32) {
    v16bf a = load_a_tile(z0n, DU, k, lane);
#pragma unroll
    for (int g = 0; g < 4; ++g)
      acc[g] = wmma_bf16(a, load_b_tile(Wi[g], DU, k, lane), acc[g]);
  }
  for (int k = 0; k < DU; k += 32) {
    v16bf a = load_a_tile(z1c, DU, k, lane);
#pragma unroll
    for (int g = 0; g < 4; ++g)
      acc[g] = wmma_bf16(a, load_b_tile(Wh[g], DU, k, lane), acc[g]);
  }
  const int n = u0 + (lane & 15);
  const int mh = (lane & 16) ? 8 : 0;
#pragma unroll
  for (int r = 0; r < 8; ++r) {
    const int m = mh + r;
    float gi = acc[0][r] + bih[n]          + bhh[n];
    float gf = acc[1][r] + bih[DU + n]     + bhh[DU + n];
    float gg = acc[2][r] + bih[2 * DU + n] + bhh[2 * DU + n];
    float go = acc[3][r] + bih[3 * DU + n] + bhh[3 * DU + n];
    const long idx = (long)m * DU + n;
    float cn = sigf(gf) * c1[idx] + sigf(gi) * tanhf(gg);
    c1[idx] = cn;
    bf16_t h = (bf16_t)(sigf(go) * tanhf(cn));
    z1n[idx] = h;
    zall_step[idx] = h;
  }
}

// ---------------- fused output GEMM + log-softmax NLL (per step) ----------------
__global__ void k_out_loss(const bf16_t* __restrict__ zall, const bf16_t* __restrict__ Wout,
                           const float* __restrict__ bout, const int* __restrict__ ys_pad,
                           float* __restrict__ partial) {
  const int s = blockIdx.x;
  const int tid = threadIdx.x;
  const int wid = tid >> 5, lane = tid & 31;
  __shared__ int   s_y[BB];
  __shared__ float s_wm[8][BB], s_ws[8][BB], s_wt[8][BB];
  __shared__ float s_nll[BB];
  if (tid < BB) s_y[tid] = (s < LM) ? ys_pad[tid * LM + s] : (OD - 1);
  __syncthreads();
  const bf16_t* A = zall + (long)s * BB * DU;
  const int mh = (lane & 16) ? 8 : 0;
  float om[8], os[8], tv[8];
#pragma unroll
  for (int r = 0; r < 8; ++r) { om[r] = NEG_INF; os[r] = 0.f; tv[r] = NEG_INF; }
  for (int nt = wid; nt < ODP / 16; nt += 8) {
    v8f acc = vzero8();
    const bf16_t* Wp = Wout + (long)nt * 16 * DU;
    for (int k = 0; k < DU; k += 32)
      acc = wmma_bf16(load_a_tile(A, DU, k, lane), load_b_tile(Wp, DU, k, lane), acc);
    const int n_abs = nt * 16 + (lane & 15);
    const float bias = (n_abs < OD) ? bout[n_abs] : 0.f;
#pragma unroll
    for (int r = 0; r < 8; ++r) {
      float v = (n_abs < OD) ? (acc[r] + bias) : NEG_INF;
      if (n_abs == s_y[mh + r]) tv[r] = fmaxf(tv[r], v);
      if (v > om[r]) { os[r] = os[r] * __expf(om[r] - v) + 1.f; om[r] = v; }
      else if (v > NEG_INF) os[r] += __expf(v - om[r]);
    }
  }
  // butterfly combine across the 16 lanes of each half-wave
#pragma unroll
  for (int r = 0; r < 8; ++r) {
    float m1 = om[r], s1 = os[r], t1 = tv[r];
    for (int d = 1; d < 16; d <<= 1) {
      float m2 = __shfl_xor(m1, d, 32);
      float s2 = __shfl_xor(s1, d, 32);
      float t2 = __shfl_xor(t1, d, 32);
      lse_merge(m1, s1, m2, s2);
      t1 = fmaxf(t1, t2);
    }
    om[r] = m1; os[r] = s1; tv[r] = t1;
  }
  if ((lane & 15) == 0) {
#pragma unroll
    for (int r = 0; r < 8; ++r) {
      const int row = mh + r;
      s_wm[wid][row] = om[r]; s_ws[wid][row] = os[r]; s_wt[wid][row] = tv[r];
    }
  }
  __syncthreads();
  if (tid < BB) {
    float m1 = NEG_INF, s1 = 0.f, t1 = NEG_INF;
    for (int w = 0; w < 8; ++w) {
      lse_merge(m1, s1, s_wm[w][tid], s_ws[w][tid]);
      t1 = fmaxf(t1, s_wt[w][tid]);
    }
    s_nll[tid] = (m1 + logf(s1)) - t1;
  }
  __syncthreads();
  if (tid == 0) {
    float acc = 0.f;
    for (int i = 0; i < BB; ++i) acc += s_nll[i];
    partial[s] = acc;
  }
}

__global__ void k_finalize(const float* __restrict__ partial, float* __restrict__ out) {
  __shared__ float sh[128];
  float a = 0.f;
  for (int i = threadIdx.x; i < ST; i += 128) a += partial[i];
  sh[threadIdx.x] = a;
  __syncthreads();
  for (int d = 64; d > 0; d >>= 1) {
    if (threadIdx.x < d) sh[threadIdx.x] += sh[threadIdx.x + d];
    __syncthreads();
  }
  if (threadIdx.x == 0) out[0] = sh[0] * ((float)LM / (float)(BB * ST));
}

// ---------------- launch ----------------
extern "C" void kernel_launch(void* const* d_in, const int* in_sizes, int n_in,
                              void* d_out, int out_size, void* d_ws, size_t ws_size,
                              hipStream_t stream) {
  (void)in_sizes; (void)n_in; (void)out_size; (void)ws_size;
  const float* hs_pad = (const float*)d_in[0];
  const int*   hlens  = (const int*)d_in[1];
  const int*   ys_pad = (const int*)d_in[2];
  const float* embed  = (const float*)d_in[3];
  const float* W_ih0  = (const float*)d_in[4];
  const float* W_hh0  = (const float*)d_in[5];
  const float* b_ih0  = (const float*)d_in[6];
  const float* b_hh0  = (const float*)d_in[7];
  const float* W_ih1  = (const float*)d_in[8];
  const float* W_hh1  = (const float*)d_in[9];
  const float* b_ih1  = (const float*)d_in[10];
  const float* b_hh1  = (const float*)d_in[11];
  const float* W_enc  = (const float*)d_in[12];
  const float* b_enc  = (const float*)d_in[13];
  const float* W_dec  = (const float*)d_in[14];
  const float* gvec   = (const float*)d_in[15];
  const float* W_out  = (const float*)d_in[16];
  const float* b_out  = (const float*)d_in[17];

  char* base = (char*)d_ws;
  size_t off = 0;
  auto alloc = [&](size_t bytes) -> char* {
    char* p = base + off;
    off = (off + bytes + 255) & ~(size_t)255;
    return p;
  };
  bf16_t* hs_bf   = (bf16_t*)alloc((size_t)BB * TT * EP * 2);
  bf16_t* Wenc_bf = (bf16_t*)alloc((size_t)AT * EP * 2);
  bf16_t* Wdec_bf = (bf16_t*)alloc((size_t)AT * DU * 2);
  bf16_t* Wih0_bf = (bf16_t*)alloc((size_t)4 * DU * (DU + EP) * 2);
  bf16_t* Whh0_bf = (bf16_t*)alloc((size_t)4 * DU * DU * 2);
  bf16_t* Wih1_bf = (bf16_t*)alloc((size_t)4 * DU * DU * 2);
  bf16_t* Whh1_bf = (bf16_t*)alloc((size_t)4 * DU * DU * 2);
  bf16_t* emb_bf  = (bf16_t*)alloc((size_t)OD * DU * 2);
  bf16_t* Wout_bf = (bf16_t*)alloc((size_t)ODP * DU * 2);
  float*  pre_enc = (float*) alloc((size_t)BB * TT * AT * 4);
  bf16_t* ey_all  = (bf16_t*)alloc((size_t)ST * BB * DU * 2);
  float*  decp    = (float*) alloc((size_t)BB * AT * 4);
  bf16_t* attc    = (bf16_t*)alloc((size_t)BB * EP * 2);
  bf16_t* z0a     = (bf16_t*)alloc((size_t)BB * DU * 2);
  bf16_t* z0b     = (bf16_t*)alloc((size_t)BB * DU * 2);
  bf16_t* z1a     = (bf16_t*)alloc((size_t)BB * DU * 2);
  bf16_t* z1b     = (bf16_t*)alloc((size_t)BB * DU * 2);
  float*  c0      = (float*) alloc((size_t)BB * DU * 4);
  float*  c1      = (float*) alloc((size_t)BB * DU * 4);
  bf16_t* zall    = (bf16_t*)alloc((size_t)ST * BB * DU * 2);
  float*  nllp    = (float*) alloc((size_t)ST * 4);

  auto cvt = [&](const float* s, bf16_t* d, int n) {
    k_cvt<<<(n + 255) / 256, 256, 0, stream>>>(s, d, n);
  };
  cvt(hs_pad, hs_bf, BB * TT * EP);
  cvt(W_enc,  Wenc_bf, AT * EP);
  cvt(W_dec,  Wdec_bf, AT * DU);
  cvt(W_ih0,  Wih0_bf, 4 * DU * (DU + EP));
  cvt(W_hh0,  Whh0_bf, 4 * DU * DU);
  cvt(W_ih1,  Wih1_bf, 4 * DU * DU);
  cvt(W_hh1,  Whh1_bf, 4 * DU * DU);
  cvt(embed,  emb_bf,  OD * DU);
  k_wout_pad<<<(ODP * DU + 255) / 256, 256, 0, stream>>>(W_out, Wout_bf);

  // zero initial recurrent state (ws is poisoned by the harness)
  k_zero32<<<(BB * DU * 2 / 4 + 255) / 256, 256, 0, stream>>>((unsigned int*)z0a, BB * DU * 2 / 4);
  k_zero32<<<(BB * DU * 2 / 4 + 255) / 256, 256, 0, stream>>>((unsigned int*)z1a, BB * DU * 2 / 4);
  k_zero32<<<(BB * DU + 255) / 256, 256, 0, stream>>>((unsigned int*)c0, BB * DU);
  k_zero32<<<(BB * DU + 255) / 256, 256, 0, stream>>>((unsigned int*)c1, BB * DU);

  k_pre_enc<<<dim3(AT / 16, (BB * TT) / 16, 1), 32, 0, stream>>>(hs_bf, Wenc_bf, b_enc, pre_enc);
  k_embed_gather<<<ST * BB, 256, 0, stream>>>(embed, ys_pad, ey_all);

  bf16_t* z0buf[2] = { z0a, z0b };
  bf16_t* z1buf[2] = { z1a, z1b };
  for (int s = 0; s < ST; ++s) {
    bf16_t* z0c = z0buf[s & 1];
    bf16_t* z0n = z0buf[(s & 1) ^ 1];
    bf16_t* z1c = z1buf[s & 1];
    bf16_t* z1n = z1buf[(s & 1) ^ 1];
    k_dec_proj<<<AT / 16, 32, 0, stream>>>(z0c, Wdec_bf, decp);
    k_attention<<<BB, 256, 0, stream>>>(pre_enc, decp, gvec, hs_pad, hlens, attc);
    k_lstm0<<<DU / 16, 32, 0, stream>>>(ey_all + (size_t)s * BB * DU, attc, z0c,
                                        Wih0_bf, Whh0_bf, b_ih0, b_hh0, c0, z0n);
    k_lstm1<<<DU / 16, 32, 0, stream>>>(z0n, z1c, Wih1_bf, Whh1_bf, b_ih1, b_hh1,
                                        c1, z1n, zall + (size_t)s * BB * DU);
  }

  k_out_loss<<<ST, 256, 0, stream>>>(zall, Wout_bf, b_out, ys_pad, nllp);
  k_finalize<<<1, 128, 0, stream>>>(nllp, (float*)d_out);
}
